// PredictionLoss_24257975288515
// MI455X (gfx1250) — matrix operand order repaired and checked by reading, not compiled, not run on hardware
//
#include <hip/hip_runtime.h>
#include <hip/hip_bf16.h>

// ---------------- constants from the reference ----------------
#define W_PRED_C 1.0f
#define W_SURP_C 0.1f
#define W_CONT_C 0.1f
#define INV_TEMP 10.0f   // 1/TEMP
#define MIN_HIST 2

static constexpr int Bc = 64;    // batch (conversations)
static constexpr int Tc = 1024;  // turns
static constexpr int Ec = 64;    // emotion classes
static constexpr int Dc = 256;   // speaker-context dim
// packed row stride: [hi 256 | lo 256] bf16 per (b,t) row
static constexpr int RS = 512;   // ushorts per packed row (1 KB)

typedef __attribute__((ext_vector_type(16))) __bf16 v16bf;
typedef __attribute__((ext_vector_type(8)))  float  v8f;
typedef __attribute__((ext_vector_type(4)))  unsigned int u32x4;
typedef __attribute__((ext_vector_type(8)))  unsigned int u32x8;

union Frag { uint4 u[2]; v16bf v; };

__device__ __forceinline__ unsigned short f2bf(float x) {
  unsigned u = __float_as_uint(x);
  unsigned r = (u + 0x7FFFu + ((u >> 16) & 1u)) >> 16;  // round-to-nearest-even
  return (unsigned short)r;
}
__device__ __forceinline__ float bf2f(unsigned short h) {
  return __uint_as_float(((unsigned)h) << 16);
}

// ---- Tensor Data Mover: 1-D 16KB DMA global -> LDS (D# per ISA 08_async_tensor) ----
// data_size = 8B units; tile = tensor = 2048 units (16KB), one row.
__device__ __forceinline__ void tdm_load_16k(unsigned lds_off, const unsigned short* gsrc) {
  unsigned long long ga = (unsigned long long)(uintptr_t)gsrc;
  u32x4 g0;
  g0.x = 1u;                                    // count=1, user descriptor
  g0.y = lds_off;                               // lds_addr (bytes, wave-relative)
  g0.z = (unsigned)ga;                          // global_addr[31:0]
  g0.w = (unsigned)(ga >> 32) | 0x80000000u;    // global_addr[56:32] | type=2 ("image")
  u32x8 g1;
  g1.s0 = 0x00030000u;                          // workgroup_mask=0, data_size=3 (8B)
  g1.s1 = 2048u << 16;                          // tensor_dim0[15:0] = 2048
  g1.s2 = 1u << 16;                             // tensor_dim0 hi=0 | tensor_dim1 = 1
  g1.s3 = 2048u << 16;                          // tensor_dim1 hi=0 | tile_dim0 = 2048
  g1.s4 = 1u;                                   // tile_dim1 = 1, tile_dim2 = 0
  g1.s5 = 2048u;                                // tensor_dim0_stride lo
  g1.s6 = 0u;
  g1.s7 = 0u;
  asm volatile("tensor_load_to_lds %0, %1" :: "s"(g0), "s"(g1) : "memory");
}

// ---------------- kernel 0: zero the small accumulator region ----------------
__global__ void zero_acc_kernel(float* acc) { acc[threadIdx.x] = 0.0f; }

// ---------------- kernel 1: CE loss + surprise + validity meta ----------------
// one wave (32 lanes) per (b,t) row of 64 logits
__global__ void __launch_bounds__(256)
pred_loss_kernel(const float* __restrict__ logits,
                 const int*   __restrict__ emo,
                 const float* __restrict__ surp,
                 const int*   __restrict__ spk,
                 int*         __restrict__ meta,
                 float*       __restrict__ acc) {
  __shared__ float part[8][3];
  const int wave = threadIdx.x >> 5;
  const int lane = threadIdx.x & 31;
  const int rid  = blockIdx.x * 8 + wave;      // [0, B*T)
  const int t    = rid & (Tc - 1);
  const float* row = logits + (size_t)rid * Ec;

  float2 x = ((const float2*)row)[lane];       // 2 logits per lane
  float m = fmaxf(x.x, x.y);
  #pragma unroll
  for (int o = 16; o; o >>= 1) m = fmaxf(m, __shfl_xor(m, o, 32));
  float s = __expf(x.x - m) + __expf(x.y - m);
  #pragma unroll
  for (int o = 16; o; o >>= 1) s += __shfl_xor(s, o, 32);

  if (lane == 0) {
    const int label = emo[rid];
    const bool valid = (label >= 0) && (t >= MIN_HIST);
    float lp = 0.f, ls = 0.f, n = 0.f;
    if (valid) {
      float pl = m + __logf(s) - row[label];   // -log_softmax at label
      lp = pl;
      ls = surp[rid] * (1.0f - __expf(-pl));
      n  = 1.0f;
    }
    meta[rid] = valid ? spk[rid] : -1;
    part[wave][0] = lp; part[wave][1] = ls; part[wave][2] = n;
  }
  __syncthreads();
  if (threadIdx.x == 0) {
    float lp = 0.f, ls = 0.f, n = 0.f;
    for (int w = 0; w < 8; ++w) { lp += part[w][0]; ls += part[w][1]; n += part[w][2]; }
    atomicAdd(&acc[0], lp);
    atomicAdd(&acc[1], ls);
    atomicAdd(&acc[2], n);
  }
}

// ---------------- kernel 2: normalize rows + split into packed bf16 hi|lo ----------------
// one wave per (b,t) row of 256 floats; output row layout: [hi0..hi255 | lo0..lo255]
__global__ void __launch_bounds__(256)
norm_split_kernel(const float* __restrict__ ctx,
                  unsigned short* __restrict__ vpk) {
  const int wave = threadIdx.x >> 5;
  const int lane = threadIdx.x & 31;
  const int rid  = blockIdx.x * 8 + wave;
  const float4* row = (const float4*)(ctx + (size_t)rid * Dc);
  float4 a = row[lane];        // dims [4*lane, +4)
  float4 b = row[32 + lane];   // dims [128+4*lane, +4)
  float ss = a.x*a.x + a.y*a.y + a.z*a.z + a.w*a.w
           + b.x*b.x + b.y*b.y + b.z*b.z + b.w*b.w;
  #pragma unroll
  for (int o = 16; o; o >>= 1) ss += __shfl_xor(ss, o, 32);
  const float rn = 1.0f / fmaxf(sqrtf(ss), 1e-12f);

  float v[8] = { a.x*rn, a.y*rn, a.z*rn, a.w*rn,
                 b.x*rn, b.y*rn, b.z*rn, b.w*rn };
  const size_t base = (size_t)rid * RS;
  #pragma unroll
  for (int c = 0; c < 2; ++c) {
    const int k = c * 4;
    unsigned short h0 = f2bf(v[k+0]), h1 = f2bf(v[k+1]);
    unsigned short h2 = f2bf(v[k+2]), h3 = f2bf(v[k+3]);
    unsigned short l0 = f2bf(v[k+0] - bf2f(h0)), l1 = f2bf(v[k+1] - bf2f(h1));
    unsigned short l2 = f2bf(v[k+2] - bf2f(h2)), l3 = f2bf(v[k+3] - bf2f(h3));
    uint2 ph, pl;
    ph.x = (unsigned)h0 | ((unsigned)h1 << 16);
    ph.y = (unsigned)h2 | ((unsigned)h3 << 16);
    pl.x = (unsigned)l0 | ((unsigned)l1 << 16);
    pl.y = (unsigned)l2 | ((unsigned)l3 << 16);
    const size_t dof = (size_t)(c ? 128 : 0) + 4 * lane;
    *(uint2*)(vpk + base + dof)       = ph;   // hi half of the row
    *(uint2*)(vpk + base + 256 + dof) = pl;   // lo half of the row
  }
}

// ---------------- kernel 3: contrastive loss, TDM-staged LDS + bf16 WMMA ----------------
// block = 4 waves; each wave owns one 16-row i-tile; the block shares each 16-column
// B panel (16 rows x (hi|lo) = 16KB) staged into LDS by wave 0 via tensor_load_to_lds,
// double-buffered on TENSORcnt + workgroup barriers.
__global__ void __launch_bounds__(128)
contrastive_kernel(const unsigned short* __restrict__ vpk,
                   const int* __restrict__ meta,
                   float* __restrict__ cnum,
                   float* __restrict__ ccnt) {
  __shared__ unsigned short sB[2][16 * RS];    // 2 x 16KB panels

  const int wave = threadIdx.x >> 5;
  const int lane = threadIdx.x & 31;
  const int half = lane >> 4;
  const int m    = lane & 15;

  const int b     = blockIdx.x >> 4;                 // 16 blocks per batch
  const int itile = ((blockIdx.x & 15) << 2) + wave; // 64 i-tiles per batch
  const int i0    = itile << 4;

  const int*            bmeta  = meta + b * Tc;
  const unsigned short* gpanel = vpk + (size_t)b * Tc * RS;
  const uint4*          Pg     = (const uint4*)gpanel;  // 8 bf16 per uint4, 64 per row

  // --- preload A fragments (rows i0..i0+15) for all 8 K-steps, hi and lo ---
  // A 16x32 bf16 layout: lanes0-15 hold K {k0..k0+7, k0+16..k0+23}; lanes16-31 +8.
  Frag Ahi[8], Alo[8];
  const int rowA = i0 + m;
  #pragma unroll
  for (int ks = 0; ks < 8; ++ks) {
    const int ib = rowA * 64 + ks * 4;           // uint4 index of hi half of row
    Ahi[ks].u[0] = Pg[ib + half];
    Ahi[ks].u[1] = Pg[ib + 2 + half];
    Alo[ks].u[0] = Pg[ib + 32 + half];           // lo half sits 32 uint4 later
    Alo[ks].u[1] = Pg[ib + 32 + 2 + half];
  }

  int rmeta[8];
  #pragma unroll
  for (int g = 0; g < 8; ++g) rmeta[g] = bmeta[i0 + g + half * 8];

  float lacc[8], ps[8], pcf[8];
  #pragma unroll
  for (int g = 0; g < 8; ++g) { lacc[g] = 0.f; ps[g] = 0.f; pcf[g] = 0.f; }

  const unsigned ldsA[2] = { (unsigned)(uintptr_t)&sB[0][0],
                             (unsigned)(uintptr_t)&sB[1][0] };
  if (wave == 0) tdm_load_16k(ldsA[0], gpanel);  // stage j-tile 0

  for (int jt = 0; jt < 64; ++jt) {
    const int buf = jt & 1;
    if (wave == 0) {
      if (jt < 63) {
        tdm_load_16k(ldsA[buf ^ 1], gpanel + (size_t)(jt + 1) * 16 * RS);
        __builtin_amdgcn_s_wait_tensorcnt(1);    // current panel done; next in flight
      } else {
        __builtin_amdgcn_s_wait_tensorcnt(0);
      }
    }
    __syncthreads();                              // panel[buf] visible to all waves

    const int j0  = jt << 4;
    const int col = j0 + m;             // this lane's column (both halves: N = lane&15)
    const int cm  = bmeta[col];
    const bool cmv = cm >= 0;

    v8f c = {0.f, 0.f, 0.f, 0.f, 0.f, 0.f, 0.f, 0.f};
    const uint4* L = (const uint4*)&sB[buf][0];   // local panel, 64 uint4 per row
    #pragma unroll
    for (int ks = 0; ks < 8; ++ks) {
      // B 32x16 bf16 layout: lanes0-15 hold K {k0..k0+7, k0+16..k0+23}; lanes16-31 +8.
      Frag Bh, Bl;
      const int ib = m * 64 + ks * 4 + half;
      Bh.u[0] = L[ib];          Bh.u[1] = L[ib + 2];
      Bl.u[0] = L[ib + 32];     Bl.u[1] = L[ib + 32 + 2];
      c = __builtin_amdgcn_wmma_f32_16x16x32_bf16(false, Ahi[ks].v, false, Bh.v,
                                                  (short)0, c, false, false);
      c = __builtin_amdgcn_wmma_f32_16x16x32_bf16(false, Ahi[ks].v, false, Bl.v,
                                                  (short)0, c, false, false);
      c = __builtin_amdgcn_wmma_f32_16x16x32_bf16(false, Alo[ks].v, false, Bh.v,
                                                  (short)0, c, false, false);
    }

    // per-lane streaming update: |sim| <= 10, so fixed-shift logsumexp (no online max)
    #pragma unroll
    for (int g = 0; g < 8; ++g) {
      const float sim = c[g] * INV_TEMP;
      const int   rg  = i0 + g + half * 8;          // global row of this C element
      const bool  dm  = cmv && (rmeta[g] >= 0) && (col != rg);
      if (dm) {
        lacc[g] += __expf(sim - 10.0f);
        if (cm == rmeta[g]) { ps[g] += sim; pcf[g] += 1.0f; }
      }
    }
    __syncthreads();                              // all waves done with panel[buf]
  }

  // reduce the 16 column-lanes of each half (rows 0-7 in lanes0-15, rows 8-15 in lanes16-31)
  #pragma unroll
  for (int g = 0; g < 8; ++g) {
    float lv = lacc[g], pv = ps[g], cv = pcf[g];
    #pragma unroll
    for (int o = 1; o < 16; o <<= 1) {
      lv += __shfl_xor(lv, o, 32);
      pv += __shfl_xor(pv, o, 32);
      cv += __shfl_xor(cv, o, 32);
    }
    lacc[g] = lv; ps[g] = pv; pcf[g] = cv;
  }

  if (m == 0) {  // lane 0 (rows 0-7) and lane 16 (rows 8-15)
    float num = 0.f, cnt = 0.f;
    #pragma unroll
    for (int g = 0; g < 8; ++g) {
      if (pcf[g] > 0.f) {
        const float log_denom = 10.0f + __logf(lacc[g]);
        num += pcf[g] * log_denom - ps[g];   // = -sum((sim - log_denom) * posf) per row
        cnt += pcf[g];
      }
    }
    atomicAdd(&cnum[b], num);
    atomicAdd(&ccnt[b], cnt);
  }
}

// ---------------- kernel 4: combine everything into the scalar loss ----------------
__global__ void finalize_kernel(const float* __restrict__ acc,
                                const float* __restrict__ cnum,
                                const float* __restrict__ ccnt,
                                float* __restrict__ out) {
  const float n  = fmaxf(acc[2], 1.0f);
  const float Lp = acc[0] / n;
  const float Ls = acc[1] / n;
  float s = 0.f, cnt = 0.f;
  for (int b = 0; b < Bc; ++b) {
    const float pc = ccnt[b];
    if (pc > 0.f) { s += cnum[b] / fmaxf(pc, 1.0f); cnt += 1.0f; }
  }
  const float Lc = s / fmaxf(cnt, 1.0f);
  out[0] = W_PRED_C * Lp + W_SURP_C * Ls + W_CONT_C * Lc;
}

// ---------------- host launcher ----------------
extern "C" void kernel_launch(void* const* d_in, const int* in_sizes, int n_in,
                              void* d_out, int out_size, void* d_ws, size_t ws_size,
                              hipStream_t stream) {
  const float* logits = (const float*)d_in[0];   // [B,T,E] f32
  const int*   emo    = (const int*)  d_in[1];   // [B,T]   i32
  const float* surp   = (const float*)d_in[2];   // [B,T]   f32
  const float* ctx    = (const float*)d_in[3];   // [B,T,D] f32
  const int*   spk    = (const int*)  d_in[4];   // [B,T]   i32
  float* out = (float*)d_out;

  // workspace layout:
  //   [0, 1KB)         : 256 f32 accumulators (0:Lpred 1:Lsurp 2:n, 64..127:cnum, 128..191:ccnt)
  //   [1KB, 1KB+256KB) : meta[B*T] int  (valid ? speaker_id : -1)
  //   then vpk[B*T][512] bf16 (64MB): per-row packed [hi 256 | lo 256]
  char* ws = (char*)d_ws;
  float* acc  = (float*)ws;
  int*   meta = (int*)(ws + 1024);
  unsigned short* vpk = (unsigned short*)(ws + 1024 + (size_t)Bc * Tc * sizeof(int));
  float* cnum = acc + 64;
  float* ccnt = acc + 128;

  zero_acc_kernel<<<1, 256, 0, stream>>>(acc);
  pred_loss_kernel<<<Bc * Tc / 8, 256, 0, stream>>>(logits, emo, surp, spk, meta, acc);
  norm_split_kernel<<<Bc * Tc / 8, 256, 0, stream>>>(ctx, vpk);
  contrastive_kernel<<<Bc * 16, 128, 0, stream>>>(vpk, meta, cnum, ccnt);
  finalize_kernel<<<1, 1, 0, stream>>>(acc, cnum, ccnt, out);
}
